// Attention_35742717837470
// MI455X (gfx1250) — matrix operand order repaired and checked by reading, not compiled, not run on hardware
//
#include <hip/hip_runtime.h>
#include <hip/hip_bf16.h>

// MI455X / gfx1250 wave32 WMMA implementation of:
// LayerNorm -> QKV proj -> causal attention w/ relative bias -> out proj.
// GEMMs + attention use v_wmma_f32_16x16x32_f16 (f16 in, f32 accum).
// Tile staging uses CDNA5 async-to-LDS copies (ASYNCcnt) with double
// buffering in the GEMM, plus global_prefetch_b8 lookahead.

typedef __attribute__((ext_vector_type(16))) _Float16 v16h;
typedef __attribute__((ext_vector_type(8)))  _Float16 h8;
typedef __attribute__((ext_vector_type(8)))  float    v8f;

union AFrag { v16h v; h8 h2[2]; };

static __device__ __forceinline__ v8f wmma_f16(const AFrag& a, const AFrag& b, v8f c) {
  return __builtin_amdgcn_wmma_f32_16x16x32_f16(false, a.v, false, b.v,
                                                (short)0, c, false, false);
}

// Async HBM -> LDS 16-byte copy (per active lane), tracked by ASYNCcnt.
// lds  = wave-relative LDS byte address (low 32 bits of a flat shared ptr)
// goff = byte offset added to the 64-bit SGPR base (GVS addressing mode)
static __device__ __forceinline__ void async_b128(unsigned lds, unsigned goff,
                                                  const _Float16* base) {
  asm volatile("global_load_async_to_lds_b128 %0, %1, %2"
               :: "v"(lds), "v"(goff), "s"(base) : "memory");
}
static __device__ __forceinline__ void wait_async0() {
  asm volatile("s_wait_asynccnt 0x0" ::: "memory");
}

#define B_   2
#define N_   2048
#define D_   1024
#define H_   16
#define DH_  64
#define MAXREL 200

// ---------------------------------------------------------------- LayerNorm
__global__ __launch_bounds__(256) void ln_cast_k(const float* __restrict__ x,
                                                 const float* __restrict__ g,
                                                 const float* __restrict__ be,
                                                 _Float16* __restrict__ xn) {
  int row = blockIdx.x;
  const float* xr = x + (size_t)row * D_;
  float vals[4], s = 0.f, ss = 0.f;
#pragma unroll
  for (int i = 0; i < 4; ++i) {
    float v = xr[threadIdx.x + 256 * i];
    vals[i] = v; s += v; ss += v * v;
  }
#pragma unroll
  for (int off = 16; off >= 1; off >>= 1) {
    s  += __shfl_xor(s, off, 32);
    ss += __shfl_xor(ss, off, 32);
  }
  __shared__ float red[2][8];
  int wid = threadIdx.x >> 5, lane = threadIdx.x & 31;
  if (lane == 0) { red[0][wid] = s; red[1][wid] = ss; }
  __syncthreads();
  s = 0.f; ss = 0.f;
#pragma unroll
  for (int i = 0; i < 8; ++i) { s += red[0][i]; ss += red[1][i]; }
  float mu  = s * (1.f / D_);
  float var = ss * (1.f / D_) - mu * mu;
  float rs  = rsqrtf(var + 1e-5f);
#pragma unroll
  for (int i = 0; i < 4; ++i) {
    int c = threadIdx.x + 256 * i;
    xn[(size_t)row * D_ + c] = (_Float16)((vals[i] - mu) * rs * g[c] + be[c]);
  }
}

// ---------------------------------------------------------------- f32->f16
__global__ __launch_bounds__(256) void cast_k(const float* __restrict__ s,
                                              _Float16* __restrict__ d, int n) {
  int i = blockIdx.x * 256 + threadIdx.x;
  if (i < n) d[i] = (_Float16)s[i];
}

// ------------------------------------------------------- generic WMMA GEMM
// C[M,Ncols] = A[M,K] * B[K,Ncols]; block tile 128x128, BK=32, 8 waves (2x4),
// wave tile 64x32 (4x2 WMMA accumulators). Double-buffered LDS: A staged via
// async-to-LDS (b128/lane), B via coalesced b128 loads + transposed stores.
// MODE 0: scatter into q/k/v [B,H,N,DH] f16 (q scaled by DH^-0.5)
// MODE 1: f32 out + bias
static __device__ __forceinline__ void gemm_stage(
    _Float16 (*As)[40], _Float16 (*Bs)[40],
    const _Float16* __restrict__ A, const _Float16* __restrict__ Bw,
    int K, int Ncols, int bm, int bn, int kb, int tid) {
  // A tile: 128 rows x 32 halves, 2 async b128 per thread
  int row = tid >> 1, ch = (tid & 1) * 16;
  size_t gofA = ((size_t)(bm * 128 + row) * K + kb + ch) * sizeof(_Float16);
  async_b128((unsigned)(size_t)&As[row][ch],     (unsigned)gofA,        A);
  async_b128((unsigned)(size_t)&As[row][ch + 8], (unsigned)(gofA + 16), A);
  // B tile: 32 k-rows x 128 cols, coalesced b128 reads, transposed b16 stores
#pragma unroll
  for (int i = 0; i < 2; ++i) {
    int c = tid + 256 * i;                 // 512 chunks of 8 halves
    int kk = c >> 4, colb = (c & 15) * 8;
    h8 bv = *(const h8*)(Bw + (size_t)(kb + kk) * Ncols + (size_t)bn * 128 + colb);
#pragma unroll
    for (int e = 0; e < 8; ++e) Bs[colb + e][kk] = bv[e];
  }
}

template <int MODE>
__global__ __launch_bounds__(256) void gemm_wmma_k(
    const _Float16* __restrict__ A, const _Float16* __restrict__ Bw,
    int K, int Ncols,
    _Float16* __restrict__ qd, _Float16* __restrict__ kd, _Float16* __restrict__ vd,
    const float* __restrict__ bias, float* __restrict__ Cout) {
  __shared__ _Float16 As[2][128][40];   // [buf][row][k]
  __shared__ _Float16 Bs[2][128][40];   // [buf][col][k] (transposed stage)
  const int tid = threadIdx.x, wid = tid >> 5, lane = tid & 31;
  const int ln = lane & 15, hi = lane >> 4;
  const int wm = wid >> 2, wn = wid & 3;
  const int bm = blockIdx.y, bn = blockIdx.x;
  v8f zero = {};
  v8f acc[4][2];
#pragma unroll
  for (int m = 0; m < 4; ++m)
#pragma unroll
    for (int n = 0; n < 2; ++n) acc[m][n] = zero;

  gemm_stage(As[0], Bs[0], A, Bw, K, Ncols, bm, bn, 0, tid);
  const int nIter = K / 32;
  for (int it = 0; it < nIter; ++it) {
    const int cur = it & 1, kb = it * 32;
    wait_async0();          // this buffer's async copies have landed
    __syncthreads();        // everyone staged & done reading other buffer
    AFrag af[4], bf[2];
#pragma unroll
    for (int m = 0; m < 4; ++m) {
      int r = wm * 64 + m * 16 + ln;
      af[m].h2[0] = *(const h8*)&As[cur][r][hi * 8];
      af[m].h2[1] = *(const h8*)&As[cur][r][16 + hi * 8];
    }
#pragma unroll
    for (int n = 0; n < 2; ++n) {
      int c = wn * 32 + n * 16 + ln;
      bf[n].h2[0] = *(const h8*)&Bs[cur][c][hi * 8];
      bf[n].h2[1] = *(const h8*)&Bs[cur][c][16 + hi * 8];
    }
    if (it + 1 < nIter) {   // overlap next tile's copies with the WMMAs
      gemm_stage(As[cur ^ 1], Bs[cur ^ 1], A, Bw, K, Ncols, bm, bn, kb + 32, tid);
      if (it + 2 < nIter) {
        __builtin_prefetch(A + (size_t)(bm * 128 + (tid >> 1)) * K + kb + 64, 0, 3);
        __builtin_prefetch(Bw + (size_t)(kb + 64 + (tid >> 3)) * Ncols + (size_t)bn * 128, 0, 3);
      }
    }
#pragma unroll
    for (int m = 0; m < 4; ++m)
#pragma unroll
      for (int n = 0; n < 2; ++n)
        acc[m][n] = wmma_f16(af[m], bf[n], acc[m][n]);
  }
#pragma unroll
  for (int m = 0; m < 4; ++m)
#pragma unroll
    for (int n = 0; n < 2; ++n)
#pragma unroll
      for (int r = 0; r < 8; ++r) {
        int grow = bm * 128 + wm * 64 + m * 16 + r + 8 * hi;
        int gcol = bn * 128 + wn * 32 + n * 16 + ln;
        float v = acc[m][n][r];
        if (MODE == 0) {
          int which = gcol >> 10, cr = gcol & 1023;
          int h = cr >> 6, d = cr & 63;
          int b = grow >> 11, nn = grow & 2047;
          size_t dst = ((size_t)(b * H_ + h) * N_ + nn) * DH_ + d;
          _Float16 hv = (_Float16)(which == 0 ? v * 0.125f : v);  // 64^-0.5
          (which == 0 ? qd : which == 1 ? kd : vd)[dst] = hv;
        } else {
          Cout[(size_t)grow * Ncols + gcol] = v + bias[gcol];
        }
      }
}

// ------------------------------------------------------- flash attention
// One block = (b,h) x 128 query rows. 8 waves, 16 rows each. Stream 32-key
// blocks: S = Q K^T (+rel bias, causal mask), online softmax, O += P V.
// K block staged via async-to-LDS; V staged coalesced + transposed.
__global__ __launch_bounds__(256) void attn_k(
    const _Float16* __restrict__ q, const _Float16* __restrict__ kbuf,
    const _Float16* __restrict__ vbuf, const float* __restrict__ rel_table,
    _Float16* __restrict__ oh) {
  __shared__ _Float16 Kl[32][72];       // [key][dim]
  __shared__ _Float16 Vl[64][40];       // [dim][key] (transposed)
  __shared__ _Float16 Pl[8][16][40];    // per-wave P repack buffer
  __shared__ float rel[2 * MAXREL];
  const int tid = threadIdx.x, wid = tid >> 5, lane = tid & 31;
  const int ln = lane & 15, hi = lane >> 4;
  const int qb = blockIdx.x & 15, bh = blockIdx.x >> 4;
  const size_t base = (size_t)bh * N_ * DH_;
  const _Float16* qp = q + base;
  const _Float16* kp = kbuf + base;
  const _Float16* vp = vbuf + base;
  for (int i = tid; i < 2 * MAXREL - 1; i += 256) rel[i] = rel_table[i];

  const int qr0 = qb * 128 + wid * 16;
  AFrag qa[2];
  const _Float16* qrow = qp + (size_t)(qr0 + ln) * DH_;
#pragma unroll
  for (int p = 0; p < 2; ++p) {
    qa[p].h2[0] = *(const h8*)(qrow + p * 32 + hi * 8);
    qa[p].h2[1] = *(const h8*)(qrow + p * 32 + 16 + hi * 8);
  }
  v8f zero = {};
  v8f o[4] = {zero, zero, zero, zero};
  float mrow[8], lrow[8];
#pragma unroll
  for (int r = 0; r < 8; ++r) { mrow[r] = -1e30f; lrow[r] = 0.f; }

  const int nj = qb * 4 + 4;  // causal: only key blocks up to this Q block
  for (int jb = 0; jb < nj; ++jb) {
    const int j0 = jb * 32;
    __syncthreads();                    // previous block's reads complete
    {
      int kr = tid >> 3, ch = (tid & 7) * 8;
      // K: contiguous rows -> async b128 straight into LDS
      size_t gof = ((size_t)(j0 + kr) * DH_ + ch) * sizeof(_Float16);
      async_b128((unsigned)(size_t)&Kl[kr][ch], (unsigned)gof, kp);
      // V: coalesced b128 read, transposed b16 scatter into LDS
      h8 vv = *(const h8*)(vp + (size_t)(j0 + kr) * DH_ + ch);
#pragma unroll
      for (int e = 0; e < 8; ++e) Vl[ch + e][kr] = vv[e];
    }
    if (jb + 1 < nj) {
      __builtin_prefetch(kp + (size_t)(j0 + 32 + (tid >> 3)) * DH_, 0, 3);
      __builtin_prefetch(vp + (size_t)(j0 + 32 + (tid >> 3)) * DH_, 0, 3);
    }
    wait_async0();
    __syncthreads();
    v8f s0 = zero, s1 = zero;  // score tiles: keys [j0,j0+16), [j0+16,j0+32)
#pragma unroll
    for (int p = 0; p < 2; ++p) {
      AFrag b0, b1;
      b0.h2[0] = *(const h8*)&Kl[ln][p * 32 + hi * 8];
      b0.h2[1] = *(const h8*)&Kl[ln][p * 32 + 16 + hi * 8];
      b1.h2[0] = *(const h8*)&Kl[16 + ln][p * 32 + hi * 8];
      b1.h2[1] = *(const h8*)&Kl[16 + ln][p * 32 + 16 + hi * 8];
      s0 = wmma_f16(qa[p], b0, s0);
      s1 = wmma_f16(qa[p], b1, s1);
    }
#pragma unroll
    for (int r = 0; r < 8; ++r) {
      const int i  = qr0 + r + 8 * hi;        // query index (C-layout row)
      const int ja = j0 + ln, jc = ja + 16;   // key indices (C-layout col)
      float v0 = s0[r], v1 = s1[r];
      int d0 = i - ja; d0 = d0 > 199 ? 199 : (d0 < -199 ? -199 : d0);
      int d1 = i - jc; d1 = d1 > 199 ? 199 : (d1 < -199 ? -199 : d1);
      v0 += rel[d0 + 199];
      v1 += rel[d1 + 199];
      if (ja > i) v0 = -1e30f;                // causal mask (== tril mask)
      if (jc > i) v1 = -1e30f;
      float mx = fmaxf(v0, v1);
#pragma unroll
      for (int off = 8; off >= 1; off >>= 1) mx = fmaxf(mx, __shfl_xor(mx, off, 16));
      float mnew  = fmaxf(mrow[r], mx);
      float alpha = __expf(mrow[r] - mnew);
      mrow[r] = mnew;
      float p0 = __expf(v0 - mnew), p1 = __expf(v1 - mnew);
      float ps = p0 + p1;
#pragma unroll
      for (int off = 8; off >= 1; off >>= 1) ps += __shfl_xor(ps, off, 16);
      lrow[r] = lrow[r] * alpha + ps;
#pragma unroll
      for (int t = 0; t < 4; ++t) o[t][r] = o[t][r] * alpha;
      Pl[wid][r + 8 * hi][ln]      = (_Float16)p0;   // C-layout -> LDS
      Pl[wid][r + 8 * hi][16 + ln] = (_Float16)p1;
    }
    asm volatile("s_wait_dscnt 0" ::: "memory");  // wave-private store->load
    AFrag pf;                                     // reload as A-fragment
    pf.h2[0] = *(const h8*)&Pl[wid][ln][hi * 8];
    pf.h2[1] = *(const h8*)&Pl[wid][ln][16 + hi * 8];
#pragma unroll
    for (int t = 0; t < 4; ++t) {
      AFrag vf;
      vf.h2[0] = *(const h8*)&Vl[t * 16 + ln][hi * 8];
      vf.h2[1] = *(const h8*)&Vl[t * 16 + ln][16 + hi * 8];
      o[t] = wmma_f16(pf, vf, o[t]);
    }
  }
  const int b = bh >> 4, h = bh & 15;
#pragma unroll
  for (int r = 0; r < 8; ++r) {
    const float inv = 1.f / lrow[r];
    const int i = qr0 + r + 8 * hi;
#pragma unroll
    for (int t = 0; t < 4; ++t) {
      const int d = t * 16 + ln;
      oh[(size_t)(b * N_ + i) * (H_ * DH_) + h * DH_ + d] = (_Float16)(o[t][r] * inv);
    }
  }
}

// ---------------------------------------------------------------- launcher
extern "C" void kernel_launch(void* const* d_in, const int* in_sizes, int n_in,
                              void* d_out, int out_size, void* d_ws, size_t ws_size,
                              hipStream_t stream) {
  const float* x     = (const float*)d_in[0];
  // d_in[1] = temporal_mask (int32) -- exactly tril, applied analytically.
  const float* ln_w  = (const float*)d_in[2];
  const float* ln_b  = (const float*)d_in[3];
  const float* w_qkv = (const float*)d_in[4];
  const float* w_out = (const float*)d_in[5];
  const float* b_out = (const float*)d_in[6];
  const float* rel_t = (const float*)d_in[7];
  (void)in_sizes; (void)n_in; (void)out_size; (void)ws_size;

  char* ws = (char*)d_ws;
  size_t off = 0;
  auto alloc = [&](size_t bytes) -> void* {
    void* p = ws + off;
    off += (bytes + 255) & ~(size_t)255;
    return p;
  };
  _Float16* xnh   = (_Float16*)alloc((size_t)B_ * N_ * D_ * 2);        // 8 MB
  _Float16* wqkvh = (_Float16*)alloc((size_t)D_ * 3 * D_ * 2);         // 6 MB
  _Float16* wouth = (_Float16*)alloc((size_t)D_ * D_ * 2);             // 2 MB
  _Float16* qh    = (_Float16*)alloc((size_t)B_ * H_ * N_ * DH_ * 2);  // 8 MB
  _Float16* kh    = (_Float16*)alloc((size_t)B_ * H_ * N_ * DH_ * 2);  // 8 MB
  _Float16* vh    = (_Float16*)alloc((size_t)B_ * H_ * N_ * DH_ * 2);  // 8 MB
  _Float16* ohb   = (_Float16*)alloc((size_t)B_ * N_ * H_ * DH_ * 2);  // 8 MB

  ln_cast_k<<<B_ * N_, 256, 0, stream>>>(x, ln_w, ln_b, xnh);
  cast_k<<<(D_ * 3 * D_) / 256, 256, 0, stream>>>(w_qkv, wqkvh, D_ * 3 * D_);
  cast_k<<<(D_ * D_) / 256, 256, 0, stream>>>(w_out, wouth, D_ * D_);

  gemm_wmma_k<0><<<dim3(3 * D_ / 128, (B_ * N_) / 128), 256, 0, stream>>>(
      xnh, wqkvh, D_, 3 * D_, qh, kh, vh, nullptr, nullptr);

  attn_k<<<B_ * H_ * (N_ / 128), 256, 0, stream>>>(qh, kh, vh, rel_t, ohb);

  gemm_wmma_k<1><<<dim3(D_ / 128, (B_ * N_) / 128), 256, 0, stream>>>(
      ohb, wouth, D_, D_, nullptr, nullptr, nullptr, b_out, (float*)d_out);
}
